// AttnClassifier_41566693491423
// MI455X (gfx1250) — matrix operand order repaired and checked by reading, not compiled, not run on hardware
//
#include <hip/hip_runtime.h>
#include <hip/hip_bf16.h>

typedef __bf16 bf16_t;
typedef __attribute__((ext_vector_type(16))) __bf16 v16bf;
typedef __attribute__((ext_vector_type(8)))  __bf16 v8bf;
typedef __attribute__((ext_vector_type(8)))  float  v8f;
typedef __attribute__((ext_vector_type(4)))  int    v4i;

#define AS1 __attribute__((address_space(1)))
#define AS3 __attribute__((address_space(3)))

#define DEV static __device__ __forceinline__

#if defined(__HIP_DEVICE_COMPILE__) && __has_builtin(__builtin_amdgcn_global_load_async_to_lds_b128)
#define HAVE_ASYNC_LDS 1
#else
#define HAVE_ASYNC_LDS 0
#endif

// ---------------------------------------------------------------------------
// WMMA fragment loaders (CDNA5 v_wmma_f32_16x16x32_bf16, wave32; this code's A/B k order)
// A (16x32 MxK): lane L row M=L&15; two contiguous 8-elem chunks at
//   K = (L>>4)*16 and +8.
// B (32x16 KxN) from W (NxK row-major): lane L col N=L&15; one contiguous
//   16-elem chunk at K = (L>>4)*16.
// C/D (16x16 f32): lane L, vgpr v -> (M = v + 8*(L>>4), N = L&15)
// ---------------------------------------------------------------------------
DEV v16bf load_frag_a(const bf16_t* p) {
    v8bf lo = *(const v8bf*)(p);
    v8bf hi = *(const v8bf*)(p + 8);
    v16bf r;
#pragma unroll
    for (int i = 0; i < 8; ++i) { r[i] = lo[i]; r[i + 8] = hi[i]; }
    return r;
}

DEV v16bf load_frag_b(const bf16_t* p) {
    return *(const v16bf*)(p);
}

// Stage one contiguous 8 KB key tile (16 rows x 256 cols bf16) into LDS.
// 256 threads x 32 B each. Async global->LDS on gfx1250 when available.
DEV void stage_tile(bf16_t* dst /*LDS*/, const bf16_t* src /*global*/, int t) {
#if HAVE_ASYNC_LDS
    __builtin_amdgcn_global_load_async_to_lds_b128(
        (AS1 v4i*)(src + t * 16),
        (AS3 v4i*)(dst + t * 16), 0, 0);
    __builtin_amdgcn_global_load_async_to_lds_b128(
        (AS1 v4i*)(src + t * 16 + 8),
        (AS3 v4i*)(dst + t * 16 + 8), 0, 0);
#else
    uint4 d0 = ((const uint4*)src)[t * 2];
    uint4 d1 = ((const uint4*)src)[t * 2 + 1];
    ((uint4*)dst)[t * 2]     = d0;
    ((uint4*)dst)[t * 2 + 1] = d1;
#endif
}

DEV void wait_stage() {
#if HAVE_ASYNC_LDS
    asm volatile("s_wait_asynccnt 0" ::: "memory");
#endif
}

// ---------------------------------------------------------------------------
// f32 -> bf16 conversion
// ---------------------------------------------------------------------------
__global__ __launch_bounds__(256) void cvt_f32_bf16(const float* __restrict__ in,
                                                    bf16_t* __restrict__ out, int n) {
    int i = blockIdx.x * 256 + threadIdx.x;
    if (i < n) out[i] = (bf16_t)in[i];
}

// ---------------------------------------------------------------------------
// C[M,N] = act(A[M,K] @ W[N,K]^T + bias[N]), bf16 in/out, f32 accum.
// Register-blocked: each wave computes a 64x32 tile (4x2 WMMA tiles).
// Per K-step: 4 A-frags + 2 B-frags -> 8 WMMAs (0.75 loads/WMMA).
// ---------------------------------------------------------------------------
__global__ __launch_bounds__(256) void gemm_bf16_wmma(
    const bf16_t* __restrict__ A, const bf16_t* __restrict__ W,
    const float* __restrict__ bias, bf16_t* __restrict__ C,
    int M, int N, int K, int relu)
{
    const int lane = threadIdx.x & 31;
    const int wave = threadIdx.x >> 5;
    const int tn   = N >> 5;                      // 32-wide n wave-tiles
    const int wt   = blockIdx.x * (blockDim.x >> 5) + wave;
    if (wt >= (M >> 6) * tn) return;              // wave-uniform
    const int m0  = (wt / tn) << 6;
    const int n0  = (wt % tn) << 5;
    const int sub = lane >> 4;
    const int l16 = lane & 15;

    const bf16_t* ap = A + (size_t)(m0 + l16) * K + (sub << 4);
    const bf16_t* bp = W + (size_t)(n0 + l16) * K + (sub << 4);
    const size_t  aK16 = (size_t)16 * K;          // row-block stride

    v8f acc[4][2];
#pragma unroll
    for (int mt = 0; mt < 4; ++mt)
#pragma unroll
        for (int nt = 0; nt < 2; ++nt) acc[mt][nt] = (v8f){};

    for (int k0 = 0; k0 < K; k0 += 32) {
        v16bf a[4], b[2];
#pragma unroll
        for (int mt = 0; mt < 4; ++mt) a[mt] = load_frag_a(ap + mt * aK16 + k0);
#pragma unroll
        for (int nt = 0; nt < 2; ++nt) b[nt] = load_frag_b(bp + nt * aK16 + k0);
#pragma unroll
        for (int mt = 0; mt < 4; ++mt)
#pragma unroll
            for (int nt = 0; nt < 2; ++nt)
                acc[mt][nt] = __builtin_amdgcn_wmma_f32_16x16x32_bf16(
                    false, a[mt], false, b[nt], (short)0, acc[mt][nt], false, false);
    }

#pragma unroll
    for (int nt = 0; nt < 2; ++nt) {
        const int col = n0 + nt * 16 + l16;
        const float bv = bias ? bias[col] : 0.0f;
#pragma unroll
        for (int mt = 0; mt < 4; ++mt) {
            const int row0 = m0 + mt * 16 + (sub << 3);
#pragma unroll
            for (int v = 0; v < 8; ++v) {
                float x = acc[mt][nt][v] + bv;
                if (relu) x = fmaxf(x, 0.0f);
                C[(size_t)(row0 + v) * N + col] = (bf16_t)x;
            }
        }
    }
}

// ---------------------------------------------------------------------------
// Energy tile from LDS-staged keys: acc += q_tile(16x256) . k_tile(16x256)^T
// ---------------------------------------------------------------------------
DEV v8f energy_tile(const v16bf* afr, const bf16_t* kb /*LDS row base*/) {
    v8f acc = {};
#pragma unroll
    for (int kk = 0; kk < 8; ++kk) {
        v16bf b = *(const v16bf*)(kb + kk * 32);
        acc = __builtin_amdgcn_wmma_f32_16x16x32_bf16(false, afr[kk], false, b,
                                                      (short)0, acc, false, false);
    }
    return acc;
}

// ---------------------------------------------------------------------------
// Pass A: per-query-row softmax stats (online max/sum over all 8192 keys).
// 8 query tiles per block (1/wave); key tiles double-buffered in LDS and
// shared by all 8 waves (8x L2-traffic reduction).
// ---------------------------------------------------------------------------
__global__ __launch_bounds__(256) void attn_softmax_stats(
    const bf16_t* __restrict__ Q, const bf16_t* __restrict__ Km,
    float* __restrict__ Mrow, float* __restrict__ Zrow)
{
    __shared__ __align__(32) bf16_t kbuf[2][16 * 256];
    const int t    = threadIdx.x;
    const int lane = t & 31;
    const int wave = t >> 5;
    const int i0   = (blockIdx.x * 8 + wave) << 4;
    const int sub  = lane >> 4;
    const int l16  = lane & 15;

    v16bf afr[8];
    const bf16_t* qp = Q + (size_t)(i0 + l16) * 256 + (sub << 4);
#pragma unroll
    for (int kk = 0; kk < 8; ++kk) afr[kk] = load_frag_a(qp + kk * 32);

    float rmax[8], rsum[8];
#pragma unroll
    for (int v = 0; v < 8; ++v) { rmax[v] = -INFINITY; rsum[v] = 0.0f; }

    stage_tile(kbuf[0], Km, t);
    wait_stage();
    __syncthreads();

    for (int jt = 0; jt < 512; ++jt) {
        const int cur = jt & 1;
        if (jt + 1 < 512)
            stage_tile(kbuf[cur ^ 1], Km + (size_t)(jt + 1) * (16 * 256), t);

        v8f acc = energy_tile(afr, kbuf[cur] + l16 * 256 + (sub << 4));

#pragma unroll
        for (int v = 0; v < 8; ++v) {
            float e = acc[v];
            float tmax = e;
            tmax = fmaxf(tmax, __shfl_xor(tmax, 1, 32));
            tmax = fmaxf(tmax, __shfl_xor(tmax, 2, 32));
            tmax = fmaxf(tmax, __shfl_xor(tmax, 4, 32));
            tmax = fmaxf(tmax, __shfl_xor(tmax, 8, 32));
            float nm = fmaxf(rmax[v], tmax);
            float p  = __expf(e - nm);
            p += __shfl_xor(p, 1, 32);
            p += __shfl_xor(p, 2, 32);
            p += __shfl_xor(p, 4, 32);
            p += __shfl_xor(p, 8, 32);
            rsum[v] = rsum[v] * __expf(rmax[v] - nm) + p;
            rmax[v] = nm;
        }
        wait_stage();        // next tile fully in LDS
        __syncthreads();     // everyone done reading cur
    }

    if (l16 == 0) {
#pragma unroll
        for (int v = 0; v < 8; ++v) {
            int row = i0 + (sub << 3) + v;
            Mrow[row] = rmax[v];
            Zrow[row] = rsum[v];
        }
    }
}

// ---------------------------------------------------------------------------
// Pass B: recompute energies, accumulate column weights
//   wcol[j] += sum_i exp(e_ij - M_i) / Z_i     (f32 global atomics)
// ---------------------------------------------------------------------------
__global__ __launch_bounds__(256) void attn_colsum(
    const bf16_t* __restrict__ Q, const bf16_t* __restrict__ Km,
    const float* __restrict__ Mrow, const float* __restrict__ Zrow,
    float* __restrict__ wcol)
{
    __shared__ __align__(32) bf16_t kbuf[2][16 * 256];
    const int t    = threadIdx.x;
    const int lane = t & 31;
    const int wave = t >> 5;
    const int i0   = (blockIdx.x * 8 + wave) << 4;
    const int sub  = lane >> 4;
    const int l16  = lane & 15;

    v16bf afr[8];
    const bf16_t* qp = Q + (size_t)(i0 + l16) * 256 + (sub << 4);
#pragma unroll
    for (int kk = 0; kk < 8; ++kk) afr[kk] = load_frag_a(qp + kk * 32);

    float mi[8], zi[8];
#pragma unroll
    for (int v = 0; v < 8; ++v) {
        int row = i0 + (sub << 3) + v;
        mi[v] = Mrow[row];
        zi[v] = 1.0f / Zrow[row];
    }

    stage_tile(kbuf[0], Km, t);
    wait_stage();
    __syncthreads();

    for (int jt = 0; jt < 512; ++jt) {
        const int cur = jt & 1;
        if (jt + 1 < 512)
            stage_tile(kbuf[cur ^ 1], Km + (size_t)(jt + 1) * (16 * 256), t);

        v8f acc = energy_tile(afr, kbuf[cur] + l16 * 256 + (sub << 4));

        float cs = 0.0f;
#pragma unroll
        for (int v = 0; v < 8; ++v)
            cs += __expf(acc[v] - mi[v]) * zi[v];
        cs += __shfl_xor(cs, 16, 32);            // rows 0-7 + rows 8-15
        if (sub == 0) atomicAdd(&wcol[jt * 16 + l16], cs);

        wait_stage();
        __syncthreads();
    }
}

// ---------------------------------------------------------------------------
// Finalize: x2[c] = (1/N) * sum_n (gamma*wcol[n] + 1) * h2[n,c]
//           logits[k] = x2 . W3[k,:] + b3[k]
// ---------------------------------------------------------------------------
__global__ __launch_bounds__(256) void finalize_kernel(
    const bf16_t* __restrict__ h2, const float* __restrict__ wcol,
    const float* __restrict__ gamma, const float* __restrict__ W3,
    const float* __restrict__ b3, float* __restrict__ out)
{
    __shared__ float coef[256];
    __shared__ float red[256];
    const int c = threadIdx.x;
    const float g = gamma[0];

    float acc = 0.0f;
    for (int base = 0; base < 8192; base += 256) {
        coef[c] = g * wcol[base + c] + 1.0f;
        __syncthreads();
#pragma unroll 4
        for (int n = 0; n < 256; ++n)
            acc += coef[n] * (float)h2[(size_t)(base + n) * 256 + c];
        __syncthreads();
    }
    const float x2 = acc * (1.0f / 8192.0f);

    red[c] = x2 * W3[c];
    __syncthreads();
    for (int s = 128; s > 0; s >>= 1) { if (c < s) red[c] += red[c + s]; __syncthreads(); }
    float logit0 = red[0];
    __syncthreads();
    red[c] = x2 * W3[256 + c];
    __syncthreads();
    for (int s = 128; s > 0; s >>= 1) { if (c < s) red[c] += red[c + s]; __syncthreads(); }
    if (c == 0) { out[0] = logit0 + b3[0]; out[1] = red[0] + b3[1]; }
}

// ---------------------------------------------------------------------------
// Host orchestration
// ---------------------------------------------------------------------------
static constexpr size_t OFF_XB  = 0;            // 8192x1024 bf16 = 16 MiB
static constexpr size_t OFF_H1  = OFF_XB  + (size_t)8192 * 1024 * 2;
static constexpr size_t OFF_H2  = OFF_H1  + (size_t)8192 * 512 * 2;
static constexpr size_t OFF_QB  = OFF_H2  + (size_t)8192 * 256 * 2;
static constexpr size_t OFF_KB  = OFF_QB  + (size_t)8192 * 256 * 2;
static constexpr size_t OFF_W1B = OFF_KB  + (size_t)8192 * 256 * 2;
static constexpr size_t OFF_W2B = OFF_W1B + (size_t)512 * 1024 * 2;
static constexpr size_t OFF_WQB = OFF_W2B + (size_t)256 * 512 * 2;
static constexpr size_t OFF_WKB = OFF_WQB + (size_t)256 * 256 * 2;
static constexpr size_t OFF_MR  = OFF_WKB + (size_t)256 * 256 * 2;
static constexpr size_t OFF_ZR  = OFF_MR  + (size_t)8192 * 4;
static constexpr size_t OFF_WC  = OFF_ZR  + (size_t)8192 * 4;

extern "C" void kernel_launch(void* const* d_in, const int* in_sizes, int n_in,
                              void* d_out, int out_size, void* d_ws, size_t ws_size,
                              hipStream_t stream)
{
    (void)in_sizes; (void)n_in; (void)out_size; (void)ws_size;

    const float* x     = (const float*)d_in[0];
    const float* W1    = (const float*)d_in[1];
    const float* b1    = (const float*)d_in[2];
    const float* W2    = (const float*)d_in[3];
    const float* b2    = (const float*)d_in[4];
    const float* Wq    = (const float*)d_in[5];
    const float* bq    = (const float*)d_in[6];
    const float* Wk    = (const float*)d_in[7];
    const float* bk    = (const float*)d_in[8];
    const float* gamma = (const float*)d_in[9];
    const float* W3    = (const float*)d_in[10];
    const float* b3    = (const float*)d_in[11];
    float* out = (float*)d_out;

    char* ws = (char*)d_ws;
    bf16_t* xb  = (bf16_t*)(ws + OFF_XB);
    bf16_t* h1  = (bf16_t*)(ws + OFF_H1);
    bf16_t* h2  = (bf16_t*)(ws + OFF_H2);
    bf16_t* qb  = (bf16_t*)(ws + OFF_QB);
    bf16_t* kb  = (bf16_t*)(ws + OFF_KB);
    bf16_t* w1b = (bf16_t*)(ws + OFF_W1B);
    bf16_t* w2b = (bf16_t*)(ws + OFF_W2B);
    bf16_t* wqb = (bf16_t*)(ws + OFF_WQB);
    bf16_t* wkb = (bf16_t*)(ws + OFF_WKB);
    float*  Mr  = (float*)(ws + OFF_MR);
    float*  Zr  = (float*)(ws + OFF_ZR);
    float*  wc  = (float*)(ws + OFF_WC);

    // 1) f32 -> bf16 conversions
    cvt_f32_bf16<<<(8192 * 1024 + 255) / 256, 256, 0, stream>>>(x,  xb,  8192 * 1024);
    cvt_f32_bf16<<<(512 * 1024 + 255) / 256, 256, 0, stream>>>(W1, w1b, 512 * 1024);
    cvt_f32_bf16<<<(256 * 512 + 255) / 256, 256, 0, stream>>>(W2, w2b, 256 * 512);
    cvt_f32_bf16<<<(256 * 256 + 255) / 256, 256, 0, stream>>>(Wq, wqb, 256 * 256);
    cvt_f32_bf16<<<(256 * 256 + 255) / 256, 256, 0, stream>>>(Wk, wkb, 256 * 256);

    // 2) fc1: h1 = relu(x @ W1^T + b1)   64x32 tiles: (8192/64)*(512/32)=2048 waves
    gemm_bf16_wmma<<<2048 / 8, 256, 0, stream>>>(xb, w1b, b1, h1, 8192, 512, 1024, 1);
    // 3) fc2: h2 = relu(h1 @ W2^T + b2)  (8192/64)*(256/32)=1024 waves
    gemm_bf16_wmma<<<1024 / 8, 256, 0, stream>>>(h1, w2b, b2, h2, 8192, 256, 512, 1);
    // 4) q = h2 @ Wq^T + bq ; k = h2 @ Wk^T + bk
    gemm_bf16_wmma<<<1024 / 8, 256, 0, stream>>>(h2, wqb, bq, qb, 8192, 256, 256, 0);
    gemm_bf16_wmma<<<1024 / 8, 256, 0, stream>>>(h2, wkb, bk, kb, 8192, 256, 256, 0);

    // 5) softmax row stats (512 query tiles, 8 waves/block, LDS-staged keys)
    attn_softmax_stats<<<64, 256, 0, stream>>>(qb, kb, Mr, Zr);

    // 6) attention column sums (zero accumulator first; capture-safe memset)
    (void)hipMemsetAsync(wc, 0, 8192 * sizeof(float), stream);
    attn_colsum<<<64, 256, 0, stream>>>(qb, kb, Mr, Zr, wc);

    // 7) finalize: weighted mean + classifier head
    finalize_kernel<<<1, 256, 0, stream>>>(h2, wc, gamma, W3, b3, out);
}
